// LSTMBlockFusedCell_58188216926792
// MI455X (gfx1250) — matrix operand order repaired and checked by reading, not compiled, not run on hardware
//
#include <hip/hip_runtime.h>

// BlockLSTM fused forward for MI455X (gfx1250, wave32, WMMA).
// T=512 sequential steps; per step: z[64,2048] = [x_t|h_prev][64,1024] @ W[1024,2048] + b
// then gate pointwise. bf16 WMMA (16x16x32) with fp32 accumulation; W converted
// once per call to transposed bf16 (4 MB, L2-resident across all 512 steps).

#define T_STEPS 512
#define B_SZ 64
#define I_SZ 512
#define U_SZ 512
#define K_TOT (I_SZ + U_SZ)   // 1024
#define N_TOT (4 * U_SZ)      // 2048

typedef __bf16 bf16_t;
typedef __attribute__((ext_vector_type(16))) bf16_t v16bf;
typedef __attribute__((ext_vector_type(8)))  float  v8f;

__device__ __forceinline__ unsigned short f2bf(float f) {
  union { float f; unsigned u; } v; v.f = f;
  unsigned r = v.u + 0x7FFFu + ((v.u >> 16) & 1u);  // round-to-nearest-even
  return (unsigned short)(r >> 16);
}

// W [K_TOT, N_TOT] fp32 row-major  ->  wT [N_TOT, K_TOT] bf16 (transposed)
__global__ void convert_w_kernel(const float* __restrict__ w,
                                 unsigned short* __restrict__ wT) {
  int idx = blockIdx.x * blockDim.x + threadIdx.x;  // idx = n*1024 + k
  if (idx >= N_TOT * K_TOT) return;
  int n = idx >> 10;
  int k = idx & 1023;
  wT[idx] = f2bf(w[k * N_TOT + n]);
}

// One timestep GEMM: z = [x_t | h_prev] @ W + b.
// Grid (32, 4): blockIdx.y = 16-row band of B, blockIdx.x*4 + wave = 16-col tile of N.
// Block = 128 threads = 4 waves; each wave owns one 16x16 output tile, K-loop of 32 WMMAs.
__global__ void __launch_bounds__(128)
lstm_gemm_kernel(const float* __restrict__ x_t,
                 const float* __restrict__ h_prev,
                 const unsigned short* __restrict__ wT,
                 const float* __restrict__ bias,
                 float* __restrict__ z) {
  __shared__ unsigned short aL[16 * K_TOT];  // 32 KB: this block's 16-row A band, bf16

  const int tm   = blockIdx.y;
  const int wave = threadIdx.x >> 5;
  const int lane = threadIdx.x & 31;
  const int tn   = blockIdx.x * 4 + wave;

  // Stage A band (x_t columns 0..511, h_prev columns 512..1023) into LDS as bf16.
  for (int i = threadIdx.x; i < 16 * K_TOT; i += 128) {
    int r = i >> 10;
    int k = i & 1023;
    float v = (k < I_SZ) ? x_t[(tm * 16 + r) * I_SZ + k]
                         : h_prev[(tm * 16 + r) * U_SZ + (k - I_SZ)];
    aL[i] = f2bf(v);
  }
  __syncthreads();

  const int hi  = lane >> 4;
  const int l4  = lane & 15;
  // A fragment (16-bit A 16x32): lane<16 -> row=l4, K {0..7, 16..23};
  //                              lane>=16 -> row=l4, K {8..15, 24..31}
  const int arow  = l4 * K_TOT;
  const int aoff0 = hi ? 8 : 0;
  const int aoff1 = hi ? 24 : 16;
  // B fragment (32x16): lane<16 -> col=l4, K {0..7, 16..23}; lane>=16 -> col=l4, K {8..15, 24..31}.
  // wT is [N][K], so each of the lane's two 8-value K runs is contiguous bf16.
  const int col = tn * 16 + l4;
  const unsigned short* wcol = wT + (size_t)col * K_TOT + hi * 8;

  union Frag { uint4 q[2]; v16bf v; };
  v8f acc = {};

  #pragma unroll 4
  for (int kb = 0; kb < K_TOT / 32; ++kb) {
    Frag a, b;
    a.q[0] = *(const uint4*)&aL[arow + kb * 32 + aoff0];   // ds_load_b128
    a.q[1] = *(const uint4*)&aL[arow + kb * 32 + aoff1];   // ds_load_b128
    b.q[0] = *(const uint4*)&wcol[kb * 32];                // global_load_b128
    b.q[1] = *(const uint4*)&wcol[kb * 32 + 16];           // global_load_b128
    acc = __builtin_amdgcn_wmma_f32_16x16x32_bf16(
        false, a.v, false, b.v, (short)0, acc, false, false);
  }

  // Epilogue: add bias (per column) and store tile.
  // C/D layout: vgpr v holds row (v + 8*hi), col = lane%16.
  const float bv = bias[col];
  #pragma unroll
  for (int v = 0; v < 8; ++v) {
    int m = v + hi * 8;
    z[(size_t)(tm * 16 + m) * N_TOT + col] = acc[v] + bv;
  }
}

// Gate pointwise: TF BlockLSTM order along N is [i | ci | f | o].
__global__ void lstm_pointwise_kernel(const float* __restrict__ z,
                                      float* __restrict__ cs,
                                      float* __restrict__ h_out,
                                      float* __restrict__ final_cs,
                                      float* __restrict__ final_h,
                                      int is_last) {
  int idx = blockIdx.x * blockDim.x + threadIdx.x;  // b*U + u
  if (idx >= B_SZ * U_SZ) return;
  int b = idx >> 9;
  int u = idx & 511;
  const float* zr = z + (size_t)b * N_TOT;
  float zi  = zr[u];
  float zci = zr[U_SZ + u];
  float zf  = zr[2 * U_SZ + u];
  float zo  = zr[3 * U_SZ + u];

  float ig = 1.f / (1.f + __expf(-zi));
  float fg = 1.f / (1.f + __expf(-(zf + 1.0f)));  // FORGET_BIAS = 1.0
  float ci = tanhf(zci);
  float c  = ci * ig + cs[idx] * fg;              // in-place cs: disjoint per thread
  float og = 1.f / (1.f + __expf(-zo));
  float h  = tanhf(c) * og;

  cs[idx]   = c;
  h_out[idx] = h;
  if (is_last) {
    final_cs[idx] = c;
    final_h[idx]  = h;
  }
}

extern "C" void kernel_launch(void* const* d_in, const int* in_sizes, int n_in,
                              void* d_out, int out_size, void* d_ws, size_t ws_size,
                              hipStream_t stream) {
  const float* x   = (const float*)d_in[0];  // [T,B,I]
  const float* cs0 = (const float*)d_in[1];  // [B,U]
  const float* h0  = (const float*)d_in[2];  // [B,U]
  const float* w   = (const float*)d_in[3];  // [I+U, 4U]
  const float* b   = (const float*)d_in[4];  // [4U]
  float* out = (float*)d_out;                // [T*B*U | final_cs B*U | final_h B*U]

  char* ws = (char*)d_ws;
  unsigned short* wT = (unsigned short*)ws;                         // 4 MB
  float* z  = (float*)(ws + (size_t)N_TOT * K_TOT * 2);             // 512 KB
  float* cs = (float*)(ws + (size_t)N_TOT * K_TOT * 2
                          + (size_t)B_SZ * N_TOT * sizeof(float));  // 128 KB

  // W -> transposed bf16 (L2-resident across all 512 steps).
  convert_w_kernel<<<(N_TOT * K_TOT + 255) / 256, 256, 0, stream>>>(w, wT);
  // cs state initialized from cs0 every call (deterministic).
  hipMemcpyAsync(cs, cs0, (size_t)B_SZ * U_SZ * sizeof(float),
                 hipMemcpyDeviceToDevice, stream);

  float* final_cs = out + (size_t)T_STEPS * B_SZ * U_SZ;
  float* final_h  = final_cs + (size_t)B_SZ * U_SZ;

  for (int t = 0; t < T_STEPS; ++t) {
    const float* hp = (t == 0) ? h0 : out + (size_t)(t - 1) * B_SZ * U_SZ;
    lstm_gemm_kernel<<<dim3(32, 4), 128, 0, stream>>>(
        x + (size_t)t * B_SZ * I_SZ, hp, wT, b, z);
    lstm_pointwise_kernel<<<(B_SZ * U_SZ + 255) / 256, 256, 0, stream>>>(
        z, cs, out + (size_t)t * B_SZ * U_SZ, final_cs, final_h,
        (t == T_STEPS - 1) ? 1 : 0);
  }
}
